// MultiHeadAttention_28424093565043
// MI455X (gfx1250) — compile-verified, matrix operand order checked by reading, not run on hardware
//
#include <hip/hip_runtime.h>
#include <hip/hip_bf16.h>
#include <math.h>

#define S_LEN  4096
#define B_SZ   2
#define NHEAD  8
#define HDIM   64
#define DMODEL 512

typedef __attribute__((ext_vector_type(16))) _Float16 v16h;
typedef __attribute__((ext_vector_type(8)))  _Float16 v8h;
typedef __attribute__((ext_vector_type(8)))  float    v8f;
typedef __attribute__((ext_vector_type(4)))  int      v4i;

// ---------------- async global->LDS staging (gfx1250 ASYNCcnt path) --------
#if defined(__gfx1250__) && __has_builtin(__builtin_amdgcn_global_load_async_to_lds_b128)
#define HAS_ASYNC_LDS 1
#else
#define HAS_ASYNC_LDS 0
#endif

#if HAS_ASYNC_LDS
typedef __attribute__((address_space(1))) v4i* gp_v4i;  // global int4*
typedef __attribute__((address_space(3))) v4i* lp_v4i;  // LDS int4*
#endif

// copy 16 bytes (8 halves) from global to LDS
__device__ __forceinline__ void cp16B(_Float16* lds_dst, const _Float16* gsrc) {
#if HAS_ASYNC_LDS
  __builtin_amdgcn_global_load_async_to_lds_b128(
      (gp_v4i)(void*)(_Float16*)gsrc,
      (lp_v4i)(void*)lds_dst,
      /*offset=*/0, /*cpol=*/0);
#else
  *(v8h*)lds_dst = *(const v8h*)gsrc;
#endif
}

__device__ __forceinline__ void cp_wait() {
#if HAS_ASYNC_LDS
#if __has_builtin(__builtin_amdgcn_s_wait_asynccnt)
  __builtin_amdgcn_s_wait_asynccnt(0);
#endif
#endif
}

// ---------------- WMMA helpers ---------------------------------------------
// 16-half fragment from LDS: two 16B chunks at +0 and +16 halves, per ISA
// 16-bit A/B layout (halves 0..7 -> K = hi*8+i, halves 8..15 -> K = 16+hi*8+i)
__device__ __forceinline__ v16h ld_frag(const _Float16* p) {
  v8h lo = *(const v8h*)(p);
  v8h hh = *(const v8h*)(p + 16);
  return __builtin_shufflevector(lo, hh, 0,1,2,3,4,5,6,7,8,9,10,11,12,13,14,15);
}

__device__ __forceinline__ v8f wmma16(v16h a, v16h b, v8f c) {
  return __builtin_amdgcn_wmma_f32_16x16x32_f16(
      false, a, false, b, (short)0, c, false, false);
}

__device__ __forceinline__ float redmax16(float v) {
#pragma unroll
  for (int m = 1; m <= 8; m <<= 1) v = fmaxf(v, __shfl_xor(v, m, 32));
  return v;
}
__device__ __forceinline__ float redsum16(float v) {
#pragma unroll
  for (int m = 1; m <= 8; m <<= 1) v += __shfl_xor(v, m, 32);
  return v;
}

// ---------------------------------------------------------------- convert
__global__ void cvt_f32_f16(const float* __restrict__ in,
                            _Float16* __restrict__ out, int n) {
  int i = (blockIdx.x * blockDim.x + threadIdx.x) * 4;
  if (i + 3 < n) {
    float4 v = *(const float4*)(in + i);
    out[i + 0] = (_Float16)v.x;
    out[i + 1] = (_Float16)v.y;
    out[i + 2] = (_Float16)v.z;
    out[i + 3] = (_Float16)v.w;
  }
}

// ---------------------------------------------------------------- GEMM
// C[8192 x 512] = A(f16, ld=512) @ W(f32 [512 x 512]) + bias
// MODE 0: f16 out remapped to [b, h, s, 64] head-major (QKV path)
// MODE 1: f32 out row-major [8192 x 512] straight to d_out (Wo path)
// Block: 128 threads (4 waves); workgroup tile 128(M) x 64(N).
template <int MODE>
__global__ void gemm_proj(const _Float16* __restrict__ A,
                          const float* __restrict__ W,
                          const float* __restrict__ bias,
                          _Float16* __restrict__ outH,
                          float* __restrict__ outF) {
  __shared__ __align__(16) _Float16 Xl[128 * 32];
  __shared__ __align__(16) _Float16 Wt[64 * 32];   // [n][k] (pre-transposed)

  const int t    = threadIdx.x;
  const int lane = t & 31;
  const int wid  = t >> 5;
  const int l16  = lane & 15;
  const int hi   = (lane >> 4) & 1;
  const int row0 = blockIdx.y * 128;
  const int n0   = blockIdx.x * 64;

  v8f acc[2][4] = {};

  for (int k0 = 0; k0 < DMODEL; k0 += 32) {
    __syncthreads();
    // stage A rows (async b128): thread t copies row (row0+t), 32 halves
    {
      const _Float16* src = A + (size_t)(row0 + t) * DMODEL + k0;
#pragma unroll
      for (int j = 0; j < 4; ++j) cp16B(&Xl[t * 32 + j * 8], src + j * 8);
    }
    // stage W transposed with f32->f16 convert: Wt[n][k] = W[k0+k][n0+n]
#pragma unroll
    for (int i2 = 0; i2 < 16; ++i2) {
      int idx = i2 * 128 + t;
      int k   = idx >> 6;
      int n   = idx & 63;
      Wt[n * 32 + k] = (_Float16)W[(size_t)(k0 + k) * DMODEL + n0 + n];
    }
    cp_wait();
    __syncthreads();

    v16h af[2], bf[4];
#pragma unroll
    for (int mt = 0; mt < 2; ++mt)
      af[mt] = ld_frag(&Xl[(wid * 32 + mt * 16 + l16) * 32 + hi * 8]);
#pragma unroll
    for (int nt = 0; nt < 4; ++nt)
      bf[nt] = ld_frag(&Wt[(nt * 16 + l16) * 32 + hi * 8]);
#pragma unroll
    for (int mt = 0; mt < 2; ++mt)
#pragma unroll
      for (int nt = 0; nt < 4; ++nt)
        acc[mt][nt] = wmma16(af[mt], bf[nt], acc[mt][nt]);
  }

  // epilogue: bias + store (compile-time mode -> no spills)
#pragma unroll
  for (int nt = 0; nt < 4; ++nt) {
    int   ng = n0 + nt * 16 + l16;
    float bv = bias[ng];
#pragma unroll
    for (int mt = 0; mt < 2; ++mt) {
#pragma unroll
      for (int e = 0; e < 8; ++e) {
        int   rg = row0 + wid * 32 + mt * 16 + e + 8 * hi;
        float v  = acc[mt][nt][e] + bv;
        if (MODE == 0) {
          int bb = rg >> 12, ss = rg & 4095;
          int h = ng >> 6, hd = ng & 63;
          outH[((size_t)((bb * NHEAD + h) * S_LEN + ss)) * HDIM + hd] =
              (_Float16)v;
        } else {
          outF[(size_t)rg * DMODEL + ng] = v;
        }
      }
    }
  }
}

// ---------------------------------------------------------------- attention
// Block: 128 threads (4 waves), one (b, h, 64-row tile). Each wave owns a
// 16-row band. Loops over 64-key blocks: WMMA scores, NT-store raw (all
// blocks, pre-mask), online softmax + P@V WMMA only for causal blocks.
__global__ void attn_kernel(const _Float16* __restrict__ Qh,
                            const _Float16* __restrict__ Kh,
                            const _Float16* __restrict__ Vh,
                            float* __restrict__ raw,
                            _Float16* __restrict__ Ah) {
  __shared__ __align__(16) _Float16 Ql[64 * 64];
  __shared__ __align__(16) _Float16 Kl[64 * 64];
  __shared__ __align__(16) _Float16 Vt[64 * 64];     // [vd][key]
  __shared__ __align__(16) _Float16 Pl[4][16 * 64];  // per-wave P tile

  const int t    = threadIdx.x;
  const int lane = t & 31;
  const int wid  = t >> 5;
  const int l16  = lane & 15;
  const int hi   = (lane >> 4) & 1;

  const int ib = blockIdx.x;  // 64-row tile index
  const int h  = blockIdx.y;
  const int b  = blockIdx.z;
  const int r0 = ib * 64;

  const size_t headBase = ((size_t)(b * NHEAD + h)) * S_LEN * HDIM;
  const _Float16* Qg = Qh + headBase + (size_t)r0 * HDIM;

  // stage Q once (async): 4096 halves / 128 threads = 32 contiguous each
#pragma unroll
  for (int j = 0; j < 4; ++j)
    cp16B(&Ql[t * 32 + j * 8], Qg + t * 32 + j * 8);

  v8f   o_acc[4] = {};
  float m_r[8], l_r[8];
#pragma unroll
  for (int e = 0; e < 8; ++e) { m_r[e] = -INFINITY; l_r[e] = 0.0f; }

  float* rawRow = raw + ((size_t)(h * B_SZ + b)) * S_LEN * S_LEN;

  for (int jb = 0; jb < S_LEN / 64; ++jb) {
    const bool active = (jb <= ib);  // uniform across the workgroup
    __syncthreads();
    // stage K block [key][qd] (async b128)
    const _Float16* Kg = Kh + headBase + (size_t)(jb * 64) * HDIM;
#pragma unroll
    for (int j = 0; j < 4; ++j)
      cp16B(&Kl[t * 32 + j * 8], Kg + t * 32 + j * 8);
    if (active) {
      // stage V transposed: b128 global reads, 8 scalar LDS scatter stores
      const _Float16* Vg = Vh + headBase + (size_t)(jb * 64) * HDIM;
#pragma unroll
      for (int i2 = 0; i2 < 4; ++i2) {
        int idx8 = (i2 * 128 + t) * 8;
        int key  = idx8 >> 6;
        int vd0  = idx8 & 63;
        v8h v    = *(const v8h*)(Vg + idx8);
#pragma unroll
        for (int j = 0; j < 8; ++j) Vt[(vd0 + j) * 64 + key] = v[j];
      }
    }
    if (jb + 1 < S_LEN / 64)
      __builtin_prefetch(Kh + headBase + (size_t)((jb + 1) * 64) * HDIM, 0, 0);
    cp_wait();
    __syncthreads();

    // ---- scores: 16(rows) x 64(keys) per wave, K = 64 in two WMMA steps
    v8f sc[4] = {};
#pragma unroll
    for (int ks = 0; ks < 2; ++ks) {
      v16h aq = ld_frag(&Ql[(wid * 16 + l16) * 64 + ks * 32 + hi * 8]);
#pragma unroll
      for (int nt = 0; nt < 4; ++nt) {
        v16h bk = ld_frag(&Kl[(nt * 16 + l16) * 64 + ks * 32 + hi * 8]);
        sc[nt] = wmma16(aq, bk, sc[nt]);
      }
    }
    // scale by 1/sqrt(64) and NT-store the pre-mask raw scores (mandatory)
#pragma unroll
    for (int nt = 0; nt < 4; ++nt) {
#pragma unroll
      for (int e = 0; e < 8; ++e) {
        float v = sc[nt][e] * 0.125f;
        sc[nt][e] = v;
        int i = r0 + wid * 16 + e + 8 * hi;
        int j = jb * 64 + nt * 16 + l16;
        __builtin_nontemporal_store(v, rawRow + (size_t)i * S_LEN + j);
      }
    }

    if (active) {
      const float NEGBIG = -1.0e30f;
      float msk[4][8];
      float rmax[8], mnew[8], fac[8], rs[8];
#pragma unroll
      for (int e = 0; e < 8; ++e) {
        int   i = r0 + wid * 16 + e + 8 * hi;
        float v = NEGBIG;
#pragma unroll
        for (int nt = 0; nt < 4; ++nt) {
          int   j = jb * 64 + nt * 16 + l16;
          float s = (j <= i) ? sc[nt][e] : NEGBIG;
          msk[nt][e] = s;
          v = fmaxf(v, s);
        }
        rmax[e] = redmax16(v);
        mnew[e] = fmaxf(m_r[e], rmax[e]);
        fac[e]  = __expf(m_r[e] - mnew[e]);
      }
      // p = exp(s - mnew); stash f16 P tile in LDS (C-layout -> A-layout)
#pragma unroll
      for (int e = 0; e < 8; ++e) {
        float acc = 0.0f;
        int   rl  = e + 8 * hi;  // row within the wave's 16-row band
#pragma unroll
        for (int nt = 0; nt < 4; ++nt) {
          float p = __expf(msk[nt][e] - mnew[e]);
          acc += p;
          Pl[wid][rl * 64 + nt * 16 + l16] = (_Float16)p;
        }
        rs[e]  = redsum16(acc);
        l_r[e] = l_r[e] * fac[e] + rs[e];
        m_r[e] = mnew[e];
      }
      // rescale running output
#pragma unroll
      for (int nt = 0; nt < 4; ++nt)
#pragma unroll
        for (int e = 0; e < 8; ++e) o_acc[nt][e] *= fac[e];

      __syncthreads();  // uniform branch; orders Pl writes before frag reads

      // ---- P @ V: 16 x 64 += (16 x 64) @ (64 x 64)
#pragma unroll
      for (int ks = 0; ks < 2; ++ks) {
        v16h ap = ld_frag(&Pl[wid][l16 * 64 + ks * 32 + hi * 8]);
#pragma unroll
        for (int nt = 0; nt < 4; ++nt) {
          v16h bv = ld_frag(&Vt[(nt * 16 + l16) * 64 + ks * 32 + hi * 8]);
          o_acc[nt] = wmma16(ap, bv, o_acc[nt]);
        }
      }
    }
  }

  // finalize: O / l, store f16 row-major [b*S + s][h*64 + vd] for final GEMM
#pragma unroll
  for (int nt = 0; nt < 4; ++nt) {
#pragma unroll
    for (int e = 0; e < 8; ++e) {
      int i   = r0 + wid * 16 + e + 8 * hi;
      int col = h * HDIM + nt * 16 + l16;
      Ah[(size_t)(b * S_LEN + i) * DMODEL + col] =
          (_Float16)(o_acc[nt][e] / l_r[e]);
    }
  }
}

// ---------------------------------------------------------------- launch
extern "C" void kernel_launch(void* const* d_in, const int* in_sizes, int n_in,
                              void* d_out, int out_size, void* d_ws,
                              size_t ws_size, hipStream_t stream) {
  const float* x  = (const float*)d_in[0];
  const float* Wq = (const float*)d_in[1];
  const float* bq = (const float*)d_in[2];
  const float* Wk = (const float*)d_in[3];
  const float* bk = (const float*)d_in[4];
  const float* Wv = (const float*)d_in[5];
  const float* bv = (const float*)d_in[6];
  const float* Wo = (const float*)d_in[7];
  const float* bo = (const float*)d_in[8];

  const size_t NTOK = (size_t)B_SZ * S_LEN;          // 8192
  const size_t NXE  = NTOK * DMODEL;                 // 4,194,304 elements

  _Float16* xh = (_Float16*)d_ws;
  _Float16* Qh = xh + NXE;
  _Float16* Kh = Qh + NXE;
  _Float16* Vh = Kh + NXE;
  _Float16* Ah = Vh + NXE;

  float* outSA = (float*)d_out;                      // [B,S,D]
  float* raw   = outSA + NXE;                        // [H*B, S, S]

  // 1) fp32 -> fp16 activations
  cvt_f32_f16<<<(int)(NXE / 1024), 256, 0, stream>>>(x, xh, (int)NXE);

  // 2) QKV projections (WMMA GEMMs, head-major f16 outputs)
  dim3 gGemm(DMODEL / 64, (unsigned)(NTOK / 128));
  gemm_proj<0><<<gGemm, 128, 0, stream>>>(xh, Wq, bq, Qh, nullptr);
  gemm_proj<0><<<gGemm, 128, 0, stream>>>(xh, Wk, bk, Kh, nullptr);
  gemm_proj<0><<<gGemm, 128, 0, stream>>>(xh, Wv, bv, Vh, nullptr);

  // 3) flash attention + full raw-score writeback
  dim3 gAttn(S_LEN / 64, NHEAD, B_SZ);
  attn_kernel<<<gAttn, 128, 0, stream>>>(Qh, Kh, Vh, raw, Ah);

  // 4) output projection -> f32 self_attention
  gemm_proj<1><<<gGemm, 128, 0, stream>>>(Ah, Wo, bo, nullptr, outSA);
}
